// SparseAutoencoder_12189117186962
// MI455X (gfx1250) — matrix operand order repaired and checked by reading, not compiled, not run on hardware
//
#include <hip/hip_runtime.h>

// SparseAutoencoder for MI455X (gfx1250, wave32).
// Pipeline (all on `stream`):
//   1) cvt_f32_bf16   : x, W_enc -> bf16 copies in d_ws (RNE)
//   2) transpose_wdec : W_dec [D,F] -> W_decT [F,D] in d_ws (L2-resident, 50MB)
//   3) encoder GEMM   : v_wmma_f32_16x16x32_bf16; 128x128 block tile, 64x32 wave
//                       tile; A/B tiles double-buffered in LDS via
//                       global_load_async_to_lds_b128 (+ s_wait_asynccnt)
//   4) topk per row   : LDS-cached row, deterministic iterative argmax
//   5) decoder        : recon[b,:] = sum_k val_k * W_decT[idx_k,:]

#define B_   8192
#define D_   768
#define F_   16384
#define K_   32

typedef __attribute__((ext_vector_type(16))) __bf16 v16bf;
typedef __attribute__((ext_vector_type(8)))  __bf16 v8bf;
typedef __attribute__((ext_vector_type(8)))  float  v8f;

union FragAB { v16bf v; v8bf h[2]; };

// ---------------- 1) f32 -> bf16 conversion (round to nearest even) ----------
__global__ void sae_cvt_bf16(const float* __restrict__ src,
                             unsigned short* __restrict__ dst, long long n) {
  long long i = (long long)blockIdx.x * blockDim.x + threadIdx.x;
  long long stride = (long long)gridDim.x * blockDim.x;
  for (; i < n; i += stride) {
    unsigned u = __float_as_uint(src[i]);
    unsigned r = u + 0x7FFFu + ((u >> 16) & 1u);
    dst[i] = (unsigned short)(r >> 16);
  }
}

// ---------------- 2) W_dec [D,F] -> W_decT [F,D] ------------------------------
__global__ void sae_transpose(const float* __restrict__ src, float* __restrict__ dst) {
  __shared__ float tile[32][33];
  int fx = blockIdx.x * 32;
  int dy = blockIdx.y * 32;
  int tx = threadIdx.x, ty = threadIdx.y;   // 32 x 8
#pragma unroll
  for (int r = 0; r < 4; ++r)
    tile[ty + r * 8][tx] = src[(size_t)(dy + ty + r * 8) * F_ + fx + tx];
  __syncthreads();
#pragma unroll
  for (int r = 0; r < 4; ++r)
    dst[(size_t)(fx + ty + r * 8) * D_ + dy + tx] = tile[tx][ty + r * 8];
}

// ------------- async copy helpers (CDNA5 ASYNCcnt path) ----------------------
#define SAE_ASYNC_CP16(ldsaddr, gaddr)                                  \
  asm volatile("global_load_async_to_lds_b128 %0, %1, off"              \
               :: "v"(ldsaddr), "v"(gaddr) : "memory")

__device__ __forceinline__ void sae_wait_async0() {
#if __has_builtin(__builtin_amdgcn_s_wait_asynccnt)
  __builtin_amdgcn_s_wait_asynccnt(0);
#else
  asm volatile("s_wait_asynccnt 0x0" ::: "memory");
#endif
}

// ---------------- 3) encoder GEMM: dense[B,F] = x·W_encT + b_enc --------------
// block tile 128(M) x 128(N), 8 waves as 2(M) x 4(N), wave tile 64x32 = 4x2 frags
// A/B k-step tiles (128x32 bf16 each) staged in LDS, rows padded 64B -> 80B,
// double buffered; copies for step i+1 overlap WMMAs of step i.
#define SM_ROW    80            // 64B payload + 16B pad (bank-conflict-free)
#define SM_TILE   (128 * SM_ROW)   // 10240 B per tile
#define SM_BUF    (2 * SM_TILE)    // A + B per buffer

__global__ __launch_bounds__(256)
void sae_encode_wmma(const unsigned short* __restrict__ xb,   // [B,D] bf16
                     const unsigned short* __restrict__ wb,   // [F,D] bf16
                     const float* __restrict__ b_enc,         // [F]
                     float* __restrict__ dense) {             // [B,F]
  __shared__ __align__(16) unsigned char sm[2][SM_BUF];       // 40 KB

  const int t    = threadIdx.x;
  const int lane = t & 31;
  const int wave = t >> 5;
  const int wm = wave >> 2;        // 0..1
  const int wn = wave & 3;         // 0..3
  const int lmod = lane & 15;
  const int lhi  = lane >> 4;

  const int rowA0 = blockIdx.y * 128;            // block M origin
  const int rowB0 = blockIdx.x * 128;            // block N origin
  const int row0 = rowA0 + wm * 64;              // wave M origin (output)
  const int col0 = rowB0 + wn * 32;              // wave N origin (output)

  const unsigned sm0 = (unsigned)(unsigned long long)(const void*)&sm[0][0];

  const int kbA = (lhi << 3);      // A half-wave K base {0,8}
  const int kbB = (lhi << 3);      // B half-wave K base {0,8}

  v8f acc[4][2] = {};

  // issue async copies of the (k0) A/B tiles into buffer `buf`
  auto issue = [&](int buf, int k0) {
    const unsigned base = sm0 + (unsigned)buf * SM_BUF;
#pragma unroll
    for (int i = 0; i < 2; ++i) {
      const int chunk = t + i * 256;       // 0..511
      const int r = chunk >> 2;            // tile row 0..127
      const int c = chunk & 3;             // 16B chunk 0..3
      const unsigned ldsA = base + (unsigned)(r * SM_ROW + c * 16);
      const unsigned long long gA =
          (unsigned long long)(xb + (size_t)(rowA0 + r) * D_ + k0 + c * 8);
      SAE_ASYNC_CP16(ldsA, gA);
      const unsigned ldsB = base + (unsigned)(SM_TILE + r * SM_ROW + c * 16);
      const unsigned long long gB =
          (unsigned long long)(wb + (size_t)(rowB0 + r) * D_ + k0 + c * 8);
      SAE_ASYNC_CP16(ldsB, gB);
    }
  };

  const int nk = D_ / 32;          // 24 k-steps
  issue(0, 0);

  for (int kk = 0; kk < nk; ++kk) {
    sae_wait_async0();             // my copies (incl. prefetch) landed
    __syncthreads();               // everyone's copies visible
    if (kk + 1 < nk) issue((kk + 1) & 1, (kk + 1) * 32);

    const unsigned char* base = &sm[kk & 1][0];
    FragAB a[4], b[2];
#pragma unroll
    for (int i = 0; i < 4; ++i) {
      const unsigned char* p = base + (wm * 64 + i * 16 + lmod) * SM_ROW + 2 * kbA;
      a[i].h[0] = *(const v8bf*)(p);        // K = kbA .. kbA+7
      a[i].h[1] = *(const v8bf*)(p + 32);   // K = kbA+16 .. kbA+23
    }
#pragma unroll
    for (int j = 0; j < 2; ++j) {
      const unsigned char* p =
          base + SM_TILE + (wn * 32 + j * 16 + lmod) * SM_ROW + 2 * kbB;
      b[j].h[0] = *(const v8bf*)(p);        // K = kbB .. kbB+7
      b[j].h[1] = *(const v8bf*)(p + 32);   // K = kbB+16 .. kbB+23
    }
#pragma unroll
    for (int i = 0; i < 4; ++i)
#pragma unroll
      for (int j = 0; j < 2; ++j)
        acc[i][j] = __builtin_amdgcn_wmma_f32_16x16x32_bf16(
            false, a[i].v, false, b[j].v, (short)0, acc[i][j], false, false);
  }

  // epilogue: +bias (f32), store. C/D layout: VGPR r -> M = r + 8*lhi, N = lmod
#pragma unroll
  for (int j = 0; j < 2; ++j) {
    const int n = col0 + j * 16 + lmod;
    const float bias = b_enc[n];
#pragma unroll
    for (int i = 0; i < 4; ++i) {
      const int mrow = row0 + i * 16 + lhi * 8;
#pragma unroll
      for (int r = 0; r < 8; ++r)
        dense[(size_t)(mrow + r) * F_ + n] = acc[i][j][r] + bias;
    }
  }
}

// ---------------- 4) per-row top-K (abs), sparsify, emit indices/values ------
__global__ __launch_bounds__(256)
void sae_topk(float* __restrict__ code,        // [B,F]: dense in, sparse out
              int* __restrict__ idx_out,       // [B,K] int32 bits
              float* __restrict__ val_out) {   // [B,K] ws
  __shared__ float row[F_];
  __shared__ unsigned long long red[256];
  __shared__ int   sel_idx[K_];
  __shared__ float sel_val[K_];

  const int b = blockIdx.x;
  const int t = threadIdx.x;
  float* grow = code + (size_t)b * F_;

  unsigned long long local = 0ull;
  for (int e = t; e < F_; e += 256) {
    float v = grow[e];
    row[e] = v;
    unsigned key = __float_as_uint(fabsf(v));
    unsigned long long p =
        ((unsigned long long)key << 32) | (unsigned long long)(0xFFFFFFFFu - e);
    local = (p > local) ? p : local;
  }
  __syncthreads();

  for (int s = 0; s < K_; ++s) {
    red[t] = local;
    __syncthreads();
#pragma unroll
    for (int off = 128; off > 0; off >>= 1) {
      if (t < off) {
        unsigned long long o = red[t + off];
        if (o > red[t]) red[t] = o;
      }
      __syncthreads();
    }
    const unsigned long long win = red[0];
    const int widx = (int)(0xFFFFFFFFu - (unsigned)(win & 0xFFFFFFFFu));
    if (t == (widx & 255)) {              // owner of the winning element
      sel_idx[s] = widx;
      sel_val[s] = row[widx];
      row[widx] = 0.0f;
      unsigned long long nl = 0ull;
      for (int e = t; e < F_; e += 256) {
        unsigned key = __float_as_uint(fabsf(row[e]));
        unsigned long long p = ((unsigned long long)key << 32) |
                               (unsigned long long)(0xFFFFFFFFu - e);
        nl = (p > nl) ? p : nl;
      }
      local = nl;
    }
    __syncthreads();
  }

  for (int e = t; e < F_; e += 256) {
    float v = 0.0f;
#pragma unroll
    for (int s = 0; s < K_; ++s)
      if (sel_idx[s] == e) v = sel_val[s];
    grow[e] = v;
  }
  if (t < K_) {
    idx_out[(size_t)b * K_ + t] = sel_idx[t];
    val_out[(size_t)b * K_ + t] = sel_val[t];
  }
}

// ---------------- 5) decoder: recon[b,:] = sum_k val_k * W_decT[idx_k,:] -----
__global__ __launch_bounds__(256)
void sae_decode(const int* __restrict__ idx_in, const float* __restrict__ val_in,
                const float* __restrict__ wdecT,    // [F,D]
                float* __restrict__ recon) {        // [B,D]
  __shared__ int   sidx[K_];
  __shared__ float sval[K_];
  const int b = blockIdx.x;
  const int t = threadIdx.x;
  if (t < K_) {
    sidx[t] = idx_in[(size_t)b * K_ + t];
    sval[t] = val_in[(size_t)b * K_ + t];
  }
  __syncthreads();
  float a0 = 0.f, a1 = 0.f, a2 = 0.f;
#pragma unroll
  for (int k = 0; k < K_; ++k) {
    const float v = sval[k];
    const float* wr = wdecT + (size_t)sidx[k] * D_;
    a0 += v * wr[t];
    a1 += v * wr[t + 256];
    a2 += v * wr[t + 512];
  }
  float* rr = recon + (size_t)b * D_;
  rr[t] = a0; rr[t + 256] = a1; rr[t + 512] = a2;
}

extern "C" void kernel_launch(void* const* d_in, const int* in_sizes, int n_in,
                              void* d_out, int out_size, void* d_ws, size_t ws_size,
                              hipStream_t stream) {
  const float* x     = (const float*)d_in[0];   // [B,D]
  const float* W_enc = (const float*)d_in[1];   // [F,D]
  const float* b_enc = (const float*)d_in[2];   // [F]
  const float* W_dec = (const float*)d_in[3];   // [D,F]

  float* out   = (float*)d_out;
  float* recon = out;                                   // [B,D]
  float* code  = out + (size_t)B_ * D_;                 // [B,F]
  int*   idxo  = (int*)(out + (size_t)B_ * D_ + (size_t)B_ * F_);  // [B,K]

  char* ws = (char*)d_ws;
  unsigned short* xb    = (unsigned short*)(ws);                          // 12.6 MB
  unsigned short* wb    = (unsigned short*)(ws + (size_t)B_ * D_ * 2);    // 25.2 MB
  float*          wdecT = (float*)(ws + (size_t)B_ * D_ * 2
                                      + (size_t)F_ * D_ * 2);             // 50.3 MB
  float*          vals  = (float*)(ws + (size_t)B_ * D_ * 2
                                      + (size_t)F_ * D_ * 2
                                      + (size_t)F_ * D_ * 4);             // 1 MB

  sae_cvt_bf16<<<4096, 256, 0, stream>>>(x, xb, (long long)B_ * D_);
  sae_cvt_bf16<<<8192, 256, 0, stream>>>(W_enc, wb, (long long)F_ * D_);

  sae_transpose<<<dim3(F_ / 32, D_ / 32), dim3(32, 8), 0, stream>>>(W_dec, wdecT);

  sae_encode_wmma<<<dim3(F_ / 128, B_ / 128), 256, 0, stream>>>(xb, wb, b_enc, code);

  sae_topk<<<B_, 256, 0, stream>>>(code, idxo, vals);

  sae_decode<<<B_, 256, 0, stream>>>(idxo, vals, wdecT, recon);
}
